// CGConv_9680856285726
// MI455X (gfx1250) — compile-verified, matrix operand order checked by reading, not run on hardware
//
#include <hip/hip_runtime.h>
#include <hip/hip_fp16.h>

typedef __attribute__((ext_vector_type(16))) _Float16 v16h;
typedef __attribute__((ext_vector_type(8)))  _Float16 v8h;
typedef __attribute__((ext_vector_type(8)))  float    v8f;

#define NATOMS 100000
#define MNB    12
#define FDIM   64
#define BDIM   41
#define KD     (2*FDIM + BDIM)   /* 169 */
#define KPAD   192               /* 6 K-blocks of 32 */
#define COLS   128
#define EPSBN  1e-5f

/* workspace layout (floats) */
#define WS_SUM1 0
#define WS_SQ1  128
#define WS_A1   256
#define WS_C1   384
#define WS_SUM2 512
#define WS_SQ2  576
#define WS_A2   640
#define WS_C2   704
#define WS_TOT  768

#define LDA 200   /* padded LDS stride (halves) for A tile: conflict-free */
#define LDG 132   /* padded LDS stride (floats) for gated tile */

__global__ __launch_bounds__(256) void cg_zero_ws(float* ws) {
    int i = threadIdx.x;
    for (; i < WS_TOT; i += 256) ws[i] = 0.0f;
}

/* PASS 0: GEMM + accumulate per-channel sum / sumsq of raw gated (no bias).
   PASS 1: GEMM + BN1 affine + sigmoid*softplus + sum over M -> d_out, BN2 stats. */
template <int PASS>
__global__ __launch_bounds__(256) void cg_gemm(const int*   __restrict__ nidx,
                                               const float* __restrict__ af,
                                               const float* __restrict__ bf,
                                               const float* __restrict__ W,
                                               float*       __restrict__ ws,
                                               float*       __restrict__ outsum) {
    __shared__ alignas(16) _Float16 ldsA[16 * LDA];
    __shared__ alignas(16) float    gbuf[16 * LDG];

    const int tid   = threadIdx.x;
    const int wave  = tid >> 5;
    const int lane  = tid & 31;
    const int nloc  = lane & 15;
    const int khalf = lane >> 4;           /* 0 / 1 -> +8 K offset */
    const int col   = wave * 16 + nloc;    /* output channel 0..127 */

    /* Loop-invariant B fragments: lane holds column `col`, K pattern mirrors A. */
    v16h Bf[6];
#pragma unroll
    for (int kb = 0; kb < 6; ++kb) {
        v16h t;
#pragma unroll
        for (int h = 0; h < 16; ++h) {
            int koff = (h < 8) ? h : (h + 8);              /* 0..7, 16..23 */
            int k = kb * 32 + khalf * 8 + koff;
            float v = (k < KD) ? W[k * COLS + col] : 0.0f; /* zero-pad K */
            t[h] = (_Float16)v;
        }
        Bf[kb] = t;
    }

    float a1c = 0.0f, c1c = 0.0f;
    if (PASS == 1) { a1c = ws[WS_A1 + col]; c1c = ws[WS_C1 + col]; }

    float s_acc = 0.0f, q_acc = 0.0f;  /* PASS0 per-lane channel stats */
    float bsum = 0.0f, bsq = 0.0f;     /* PASS1 per-thread (f<64) BN2 stats */

    const int r  = tid >> 4;   /* staging row 0..15 */
    const int kk = tid & 15;   /* staging K phase   */

    for (int atom = blockIdx.x; atom < NATOMS; atom += gridDim.x) {
        /* ---- stage 16x192 f16 A tile: [self | nbr | bond | 0pad]; rows 12..15 zero ---- */
        int nb = 0;
        if (r < MNB) nb = nidx[atom * MNB + r];
#pragma unroll
        for (int i = 0; i < KPAD / 16; ++i) {
            int k = kk + 16 * i;
            float v = 0.0f;
            if (r < MNB) {
                if (k < FDIM)          v = af[atom * FDIM + k];
                else if (k < 2 * FDIM) v = af[nb * FDIM + (k - FDIM)];
                else if (k < KD)       v = bf[(atom * MNB + r) * BDIM + (k - 2 * FDIM)];
            }
            ldsA[r * LDA + k] = (_Float16)v;
        }
        __syncthreads();

        /* ---- per-wave 16x192 @ 192x16 via 6 WMMAs ---- */
        v8f acc = {};
        const _Float16* arow = &ldsA[nloc * LDA];
#pragma unroll
        for (int kb = 0; kb < 6; ++kb) {
            const _Float16* p = arow + kb * 32 + khalf * 8;
            v8h lo = *(const v8h*)(p);
            v8h hi = *(const v8h*)(p + 16);
            v16h a = __builtin_shufflevector(lo, hi, 0, 1, 2, 3, 4, 5, 6, 7,
                                                     8, 9, 10, 11, 12, 13, 14, 15);
            acc = __builtin_amdgcn_wmma_f32_16x16x32_f16(
                false, a, false, Bf[kb], (short)0, acc, false, false);
        }

        if (PASS == 0) {
            /* ghost rows produced exact zeros -> safe to sum unconditionally */
#pragma unroll
            for (int i = 0; i < 8; ++i) { float v = acc[i]; s_acc += v; q_acc += v * v; }
            __syncthreads();
        } else {
            /* affine BN1 (bias folded into c1), exchange through LDS for filter/core pairing */
#pragma unroll
            for (int i = 0; i < 8; ++i) {
                int row = khalf * 8 + i;
                gbuf[row * LDG + col] = acc[i] * a1c + c1c;
            }
            __syncthreads();
            if (tid < FDIM) {
                float s = 0.0f;
#pragma unroll
                for (int m = 0; m < MNB; ++m) {
                    float fl = gbuf[m * LDG + tid];
                    float co = gbuf[m * LDG + FDIM + tid];
                    float sig = 1.0f / (1.0f + __expf(-fl));
                    float sp  = fmaxf(co, 0.0f) + __logf(1.0f + __expf(-fabsf(co)));
                    s += sig * sp;
                }
                outsum[atom * FDIM + tid] = s;
                bsum += s; bsq += s * s;
            }
            __syncthreads();
        }
    }

    if (PASS == 0) {
        atomicAdd(&ws[WS_SUM1 + col], s_acc);
        atomicAdd(&ws[WS_SQ1 + col], q_acc);
    } else if (tid < FDIM) {
        atomicAdd(&ws[WS_SUM2 + tid], bsum);
        atomicAdd(&ws[WS_SQ2 + tid], bsq);
    }
}

__global__ __launch_bounds__(128) void cg_fold1(float* ws, const float* bias,
                                                const float* sc, const float* off) {
    int c = threadIdx.x;
    if (c >= COLS) return;
    const float cnt = (float)NATOMS * (float)MNB;
    float mean = ws[WS_SUM1 + c] / cnt;
    float var  = ws[WS_SQ1 + c] / cnt - mean * mean;
    float inv  = sc[c] * rsqrtf(var + EPSBN);
    ws[WS_A1 + c] = inv;
    ws[WS_C1 + c] = (bias[c] - mean) * inv + off[c];
}

__global__ __launch_bounds__(64) void cg_fold2(float* ws, const float* sc, const float* off) {
    int c = threadIdx.x;
    if (c >= FDIM) return;
    const float cnt = (float)NATOMS;
    float mean = ws[WS_SUM2 + c] / cnt;
    float var  = ws[WS_SQ2 + c] / cnt - mean * mean;
    float inv  = sc[c] * rsqrtf(var + EPSBN);
    ws[WS_A2 + c] = inv;
    ws[WS_C2 + c] = off[c] - mean * inv;
}

__global__ __launch_bounds__(256) void cg_final(const float* __restrict__ af,
                                                const float* __restrict__ ws,
                                                float* __restrict__ out) {
    int idx = blockIdx.x * 256 + threadIdx.x;
    if (idx >= NATOMS * FDIM) return;
    int f = idx & (FDIM - 1);
    float s = out[idx] * ws[WS_A2 + f] + ws[WS_C2 + f];  /* out held `summed` */
    float x = af[idx] + s;
    out[idx] = fmaxf(x, 0.0f) + __logf(1.0f + __expf(-fabsf(x)));
}

extern "C" void kernel_launch(void* const* d_in, const int* in_sizes, int n_in,
                              void* d_out, int out_size, void* d_ws, size_t ws_size,
                              hipStream_t stream) {
    const int*   nidx = (const int*)  d_in[0];
    const float* af   = (const float*)d_in[1];
    const float* bf   = (const float*)d_in[2];
    const float* W    = (const float*)d_in[3];
    const float* bias = (const float*)d_in[4];
    const float* s1   = (const float*)d_in[5];
    const float* o1   = (const float*)d_in[6];
    const float* s2   = (const float*)d_in[7];
    const float* o2   = (const float*)d_in[8];
    float* out = (float*)d_out;
    float* ws  = (float*)d_ws;

    const int gemm_blocks = 2048;  /* persistent grid-stride; amortizes B-fragment setup */

    cg_zero_ws<<<1, 256, 0, stream>>>(ws);
    cg_gemm<0><<<gemm_blocks, 256, 0, stream>>>(nidx, af, bf, W, ws, out);
    cg_fold1<<<1, 128, 0, stream>>>(ws, bias, s1, o1);
    cg_gemm<1><<<gemm_blocks, 256, 0, stream>>>(nidx, af, bf, W, ws, out);
    cg_fold2<<<1, 64, 0, stream>>>(ws, s2, o2);
    cg_final<<<(NATOMS * FDIM + 255) / 256, 256, 0, stream>>>(af, ws, out);
}
